// GINetReconEmbedding_4183298146467
// MI455X (gfx1250) — matrix operand order repaired and checked by reading, not compiled, not run on hardware
//
#include <hip/hip_runtime.h>
#include <math.h>

// Problem constants (from reference)
#define N_NODES 10000
#define N_EDGES 160000
#define EN      170000          // E + N self loops
#define DIM     300
#define NLAYER  5
#define NGRAPH  64
#define FDIM    512
#define HDIM    256
#define BNEPS   1e-5f
// Padded dims for WMMA GEMM (K multiple of 32, M/N multiple of 64)
#define DP      320             // padded 300
#define D2P     640             // padded 600
#define MP      10048           // padded 10000 (multiple of 64)

typedef __bf16 bf16;
typedef __attribute__((ext_vector_type(16))) __bf16 v16bf;
typedef __attribute__((ext_vector_type(8)))  float  v8f;

// ---------------------------------------------------------------- utilities
__global__ void k_zero(float* p, size_t n) {
    size_t i = (size_t)blockIdx.x * blockDim.x + threadIdx.x;
    if (i < n) p[i] = 0.0f;
}

__global__ void k_copy(const float* s, float* d, size_t n) {
    size_t i = (size_t)blockIdx.x * blockDim.x + threadIdx.x;
    if (i < n) d[i] = s[i];
}

__global__ void k_pad_bias(const float* b, float* d, int C, int Cp) {
    int i = blockIdx.x * blockDim.x + threadIdx.x;
    if (i < Cp) d[i] = (i < C) ? b[i] : 0.0f;
}

// ------------------------------------------------- WMMA fragment packing
// A-fragment element map (16-bit A 16x32, ISA 7.12.2):
//   lane 0-15: M=lane, VGPR v0-3 hold K=2v,2v+1 ; v4-7 hold K=16+2(v-4),+1
//   lane 16-31: same M, K shifted by +8
// Packed layout: frag = packed[(tile*32 + lane)*16 + e], 32B/lane, coalesced.
__device__ __forceinline__ int a_kidx(int lane, int e) {
    int v = e >> 1, par = e & 1, base = (lane >> 4) * 8;
    return (v < 4) ? (base + 2 * v + par) : (16 + base + 2 * (v - 4) + par);
}
// B-fragment element map (16-bit B 32x16): same K order as A fragment
//   lane 0-15: v0-3 K=2v,2v+1 ; v4-7 K=16+2(v-4),+1 ; lanes 16-31 K +8; N=lane%16
__device__ __forceinline__ int b_kidx(int lane, int e) {
    int v = e >> 1, par = e & 1;
    return a_kidx(lane, e);
}

// pack fp32 activation matrix [Mp x Kp] (fully padded, no guards) into
// bf16 A-fragments: tiles are (mt, kt) row-major, Ktiles = Kp/32
__global__ void k_pack_a(const float* __restrict__ src, int lda,
                         bf16* __restrict__ dst, int Ktiles, size_t total) {
    size_t i = (size_t)blockIdx.x * blockDim.x + threadIdx.x;
    if (i >= total) return;
    size_t tile = i >> 9;              // /512 elems per fragment tile
    int rem  = (int)(i & 511);
    int lane = rem >> 4, e = rem & 15;
    int mt = (int)(tile / Ktiles), kt = (int)(tile % Ktiles);
    int m = mt * 16 + (lane & 15);
    int k = kt * 32 + a_kidx(lane, e);
    dst[i] = (bf16)src[(size_t)m * lda + k];
}

// pack fp32 weight [R x C] into zero-padded bf16 B-fragments [Kp x Np]:
// tiles (kt, nt) row-major, Ntiles = Np/16
__global__ void k_pack_b(const float* __restrict__ w, bf16* __restrict__ dst,
                         int R, int C, int Ntiles, size_t total) {
    size_t i = (size_t)blockIdx.x * blockDim.x + threadIdx.x;
    if (i >= total) return;
    size_t tile = i >> 9;
    int rem  = (int)(i & 511);
    int lane = rem >> 4, e = rem & 15;
    int kt = (int)(tile / Ntiles), nt = (int)(tile % Ntiles);
    int n = nt * 16 + (lane & 15);
    int k = kt * 32 + b_kidx(lane, e);
    dst[i] = (k < R && n < C) ? (bf16)w[(size_t)k * C + n] : (bf16)0.0f;
}

// ---------------------------------------------------------------- embedding
__global__ void k_node_embed(const int* x, const float* e1, const float* e2, float* h) {
    size_t i = (size_t)blockIdx.x * blockDim.x + threadIdx.x;
    if (i >= (size_t)N_NODES * DIM) return;
    int row = (int)(i / DIM), d = (int)(i % DIM);
    h[i] = e1[x[row * 2] * DIM + d] + e2[x[row * 2 + 1] * DIM + d];
}

// -------------------------------------------- edge embed + message + scatter
// one wave per edge (incl. self loops), lanes stride the 300 feature dims
__global__ void k_edge_msg(const int* __restrict__ ei, const int* __restrict__ ea,
                           const float* __restrict__ h,
                           const float* __restrict__ e1, const float* __restrict__ e2,
                           const float* __restrict__ w3, const float* __restrict__ b3,
                           float* __restrict__ aggr) {
    int e    = blockIdx.x * 8 + (threadIdx.x >> 5);
    int lane = threadIdx.x & 31;
    if (e >= EN) return;
    int s, dst, a0, a1;
    float af[8];
    if (e < N_EDGES) {
        s   = ei[e];
        dst = ei[N_EDGES + e];
        const int* ar = ea + (size_t)e * 10;
        a0 = ar[0]; a1 = ar[1];
        #pragma unroll
        for (int k = 0; k < 8; ++k) af[k] = (float)ar[2 + k];
    } else {
        s = dst = e - N_EDGES;
        a0 = 4; a1 = 0;
        #pragma unroll
        for (int k = 0; k < 8; ++k) af[k] = 0.0f;
    }
    const float* hs = h  + (size_t)s  * DIM;
    const float* p1 = e1 + (size_t)a0 * DIM;
    const float* p2 = e2 + (size_t)a1 * DIM;
    for (int d = lane; d < DIM; d += 32) {
        float v = hs[d] + p1[d] + p2[d] + b3[d];
        #pragma unroll
        for (int k = 0; k < 8; ++k) v = fmaf(af[k], w3[k * DIM + d], v);
        atomicAdd(&aggr[(size_t)dst * DP + d], v);
    }
}

// ---------------------------------------------------------------- WMMA GEMM
// C[M,N] = act(A*B + bias) on pre-packed bf16 fragments.  No LDS, no barriers:
// per k-step each lane issues contiguous 32B loads (global_load_b128 pairs)
// of its A/B fragments, then two v_wmma_f32_16x16x32_bf16.
// Block = 256 thr (8 waves); block tile 64x64; wave tile 16 rows x 32 cols.
__global__ void k_wmma_gemm(const v16bf* __restrict__ Ap,
                            const v16bf* __restrict__ Bp,
                            const float* __restrict__ bias,
                            float* __restrict__ C, int ldc,
                            int Ktiles, int Ntiles, int relu) {
    const int lane = threadIdx.x & 31;
    const int w    = threadIdx.x >> 5;          // wave 0..7
    const int mt   = blockIdx.x * 4 + (w >> 1); // 16-row tile index
    const int nt0  = blockIdx.y * 4 + (w & 1) * 2;

    v8f c0 = {};
    v8f c1 = {};

    const v16bf* ap = Ap + (size_t)mt * Ktiles * 32 + lane;
    for (int kt = 0; kt < Ktiles; ++kt) {
        v16bf a  = ap[(size_t)kt * 32];
        v16bf b0 = Bp[((size_t)kt * Ntiles + nt0) * 32 + lane];
        v16bf b1 = Bp[((size_t)kt * Ntiles + nt0 + 1) * 32 + lane];
        c0 = __builtin_amdgcn_wmma_f32_16x16x32_bf16(false, a, false, b0,
                                                     (short)0, c0, false, false);
        c1 = __builtin_amdgcn_wmma_f32_16x16x32_bf16(false, a, false, b1,
                                                     (short)0, c1, false, false);
    }

    // C/D layout: VGPR v -> lanes 0-15 M=v, lanes 16-31 M=v+8; N=lane%16
    int gcol = nt0 * 16 + (lane & 15);
    float bv0 = bias[gcol];
    float bv1 = bias[gcol + 16];
    #pragma unroll
    for (int v = 0; v < 8; ++v) {
        int row = mt * 16 + ((lane < 16) ? v : v + 8);
        float x0 = c0[v] + bv0;
        float x1 = c1[v] + bv1;
        if (relu) { x0 = fmaxf(x0, 0.0f); x1 = fmaxf(x1, 0.0f); }
        C[(size_t)row * ldc + gcol]      = x0;
        C[(size_t)row * ldc + gcol + 16] = x1;
    }
}

// ---------------------------------------------------------------- BatchNorm
// one block per feature; training-mode (population) stats over N rows
__global__ void k_bn(const float* __restrict__ hn, int ld,
                     const float* __restrict__ g, const float* __restrict__ b,
                     float* __restrict__ hout, int relu) {
    __shared__ float ss[256], ss2[256];
    int d = blockIdx.x;
    float s = 0.0f, s2 = 0.0f;
    for (int i = threadIdx.x; i < N_NODES; i += 256) {
        float v = hn[(size_t)i * ld + d];
        s += v; s2 += v * v;
    }
    ss[threadIdx.x] = s; ss2[threadIdx.x] = s2;
    __syncthreads();
    for (int o = 128; o > 0; o >>= 1) {
        if (threadIdx.x < o) { ss[threadIdx.x] += ss[threadIdx.x + o];
                               ss2[threadIdx.x] += ss2[threadIdx.x + o]; }
        __syncthreads();
    }
    float mu  = ss[0] / (float)N_NODES;
    float var = ss2[0] / (float)N_NODES - mu * mu;
    float sc  = rsqrtf(var + BNEPS) * g[d];
    float sh  = b[d] - mu * sc;
    for (int i = threadIdx.x; i < N_NODES; i += 256) {
        float v = hn[(size_t)i * ld + d] * sc + sh;
        if (relu) v = fmaxf(v, 0.0f);
        hout[(size_t)i * DIM + d] = v;
    }
}

// ---------------------------------------------------------------- pooling
__global__ void k_count(const int* batch, float* cnt) {
    int i = blockIdx.x * blockDim.x + threadIdx.x;
    if (i < N_NODES) atomicAdd(&cnt[batch[i]], 1.0f);
}

__global__ void k_pool(const float* h, const int* batch, float* pooled) {
    size_t i = (size_t)blockIdx.x * blockDim.x + threadIdx.x;
    if (i >= (size_t)N_NODES * DIM) return;
    int row = (int)(i / DIM), d = (int)(i % DIM);
    atomicAdd(&pooled[(size_t)batch[row] * DIM + d], h[i]);
}

__global__ void k_pool_div(float* pooled, const float* cnt) {
    int i = blockIdx.x * blockDim.x + threadIdx.x;
    if (i >= NGRAPH * DIM) return;
    pooled[i] /= fmaxf(cnt[i / DIM], 1.0f);
}

// ------------------------------------------------- small head GEMMs (64 rows)
// act: 0=none, 1=softplus
__global__ void k_small_gemm(const float* __restrict__ A, const float* __restrict__ B,
                             const float* __restrict__ bias, float* __restrict__ C,
                             int M, int Nc, int K, int act) {
    int id = blockIdx.x * blockDim.x + threadIdx.x;
    if (id >= M * Nc) return;
    int m = id / Nc, n = id % Nc;
    float acc = bias[n];
    for (int k = 0; k < K; ++k) acc = fmaf(A[(size_t)m * K + k], B[(size_t)k * Nc + n], acc);
    if (act == 1) acc = (acc > 20.0f) ? acc : log1pf(expf(acc));
    C[id] = acc;
}

// ---------------------------------------------------------------- launcher
extern "C" void kernel_launch(void* const* d_in, const int* in_sizes, int n_in,
                              void* d_out, int out_size, void* d_ws, size_t ws_size,
                              hipStream_t stream) {
    (void)in_sizes; (void)n_in; (void)out_size; (void)ws_size;
    const int*   x     = (const int*)d_in[0];
    const int*   ei    = (const int*)d_in[1];
    const int*   ea    = (const int*)d_in[2];
    const int*   batch = (const int*)d_in[3];
    const float* xe1   = (const float*)d_in[4];
    const float* xe2   = (const float*)d_in[5];
    const float* mw1   = (const float*)d_in[6];
    const float* mb1   = (const float*)d_in[7];
    const float* mw2   = (const float*)d_in[8];
    const float* mb2   = (const float*)d_in[9];
    const float* ee1   = (const float*)d_in[10];
    const float* ee2   = (const float*)d_in[11];
    const float* ee3w  = (const float*)d_in[12];
    const float* ee3b  = (const float*)d_in[13];
    const float* bng   = (const float*)d_in[14];
    const float* bnb   = (const float*)d_in[15];
    const float* fw    = (const float*)d_in[16];
    const float* fb    = (const float*)d_in[17];
    const float* pw1   = (const float*)d_in[18];
    const float* pb1   = (const float*)d_in[19];
    const float* pw2   = (const float*)d_in[20];
    const float* pb2   = (const float*)d_in[21];
    const float* pw3   = (const float*)d_in[22];
    const float* pb3   = (const float*)d_in[23];
    float* out = (float*)d_out;

    // workspace bump allocator (256B aligned)
    char* wp = (char*)d_ws;
    auto alloc = [&](size_t bytes) -> char* {
        char* r = wp; wp += (bytes + 255) & ~(size_t)255; return r;
    };
    float* h       = (float*)alloc(sizeof(float) * (size_t)N_NODES * DIM);
    float* aggr    = (float*)alloc(sizeof(float) * (size_t)MP * DP);
    bf16*  a1_pk   = (bf16*) alloc(sizeof(bf16)  * (size_t)MP * DP);   // packed A, gemm1
    bf16*  w1_pk   = (bf16*) alloc(sizeof(bf16)  * (size_t)DP * D2P);  // packed B, gemm1
    bf16*  w2_pk   = (bf16*) alloc(sizeof(bf16)  * (size_t)D2P * DP);  // packed B, gemm2
    float* b1_pad  = (float*)alloc(sizeof(float) * D2P);
    float* b2_pad  = (float*)alloc(sizeof(float) * DP);
    float* t1      = (float*)alloc(sizeof(float) * (size_t)MP * D2P);
    bf16*  a2_pk   = (bf16*) alloc(sizeof(bf16)  * (size_t)MP * D2P);  // packed A, gemm2
    float* hnext   = (float*)alloc(sizeof(float) * (size_t)MP * DP);
    float* pooled  = (float*)alloc(sizeof(float) * NGRAPH * DIM);
    float* cnt     = (float*)alloc(sizeof(float) * NGRAPH);
    float* hf      = (float*)alloc(sizeof(float) * NGRAPH * FDIM);
    float* z1      = (float*)alloc(sizeof(float) * NGRAPH * HDIM);
    float* z2      = (float*)alloc(sizeof(float) * NGRAPH * HDIM);

    const size_t nhd  = (size_t)N_NODES * DIM;
    const size_t nagg = (size_t)MP * DP;
    const size_t nt1  = (size_t)MP * D2P;
    const size_t nw1  = (size_t)DP * D2P;
    const size_t nw2  = (size_t)D2P * DP;

    // node embedding
    k_node_embed<<<(unsigned)((nhd + 255) / 256), 256, 0, stream>>>(x, xe1, xe2, h);

    for (int l = 0; l < NLAYER; ++l) {
        // zero padded aggregation buffer, scatter messages into it
        k_zero<<<(unsigned)((nagg + 255) / 256), 256, 0, stream>>>(aggr, nagg);
        k_edge_msg<<<(EN + 7) / 8, 256, 0, stream>>>(
            ei, ea, h,
            ee1 + (size_t)l * 6 * DIM, ee2 + (size_t)l * 4 * DIM,
            ee3w + (size_t)l * 8 * DIM, ee3b + (size_t)l * DIM, aggr);

        // pack operands into WMMA fragment order (bf16)
        k_pack_a<<<(unsigned)((nagg + 255) / 256), 256, 0, stream>>>(
            aggr, DP, a1_pk, DP / 32, nagg);
        k_pack_b<<<(unsigned)((nw1 + 255) / 256), 256, 0, stream>>>(
            mw1 + (size_t)l * DIM * 2 * DIM, w1_pk, DIM, 2 * DIM, D2P / 16, nw1);
        k_pad_bias<<<(D2P + 255) / 256, 256, 0, stream>>>(
            mb1 + (size_t)l * 2 * DIM, b1_pad, 2 * DIM, D2P);

        // GEMM1 + ReLU : [MP,DP] x [DP,D2P]
        {
            dim3 g(MP / 64, D2P / 64);
            k_wmma_gemm<<<g, 256, 0, stream>>>((const v16bf*)a1_pk, (const v16bf*)w1_pk,
                                               b1_pad, t1, D2P, DP / 32, D2P / 16, 1);
        }

        k_pack_a<<<(unsigned)((nt1 + 255) / 256), 256, 0, stream>>>(
            t1, D2P, a2_pk, D2P / 32, nt1);
        k_pack_b<<<(unsigned)((nw2 + 255) / 256), 256, 0, stream>>>(
            mw2 + (size_t)l * 2 * DIM * DIM, w2_pk, 2 * DIM, DIM, DP / 16, nw2);
        k_pad_bias<<<(DP + 255) / 256, 256, 0, stream>>>(
            mb2 + (size_t)l * DIM, b2_pad, DIM, DP);

        // GEMM2 : [MP,D2P] x [D2P,DP]
        {
            dim3 g(MP / 64, DP / 64);
            k_wmma_gemm<<<g, 256, 0, stream>>>((const v16bf*)a2_pk, (const v16bf*)w2_pk,
                                               b2_pad, hnext, DP, D2P / 32, DP / 16, 0);
        }

        // BatchNorm (+ReLU except last layer) -> compact h [N,DIM]
        k_bn<<<DIM, 256, 0, stream>>>(hnext, DP, bng + (size_t)l * DIM, bnb + (size_t)l * DIM,
                                      h, (l < NLAYER - 1) ? 1 : 0);
    }

    // h_node output
    k_copy<<<(unsigned)((nhd + 255) / 256), 256, 0, stream>>>(h, out, nhd);

    // global mean pool
    k_zero<<<(NGRAPH * DIM + 255) / 256, 256, 0, stream>>>(pooled, (size_t)NGRAPH * DIM);
    k_zero<<<1, 256, 0, stream>>>(cnt, (size_t)NGRAPH);
    k_count<<<(N_NODES + 255) / 256, 256, 0, stream>>>(batch, cnt);
    k_pool<<<(unsigned)((nhd + 255) / 256), 256, 0, stream>>>(h, batch, pooled);
    k_pool_div<<<(NGRAPH * DIM + 255) / 256, 256, 0, stream>>>(pooled, cnt);

    // prediction head (tiny, 64 rows)
    k_small_gemm<<<(NGRAPH * FDIM + 255) / 256, 256, 0, stream>>>(pooled, fw, fb, hf, NGRAPH, FDIM, DIM, 0);
    k_small_gemm<<<(NGRAPH * HDIM + 255) / 256, 256, 0, stream>>>(hf, pw1, pb1, z1, NGRAPH, HDIM, FDIM, 1);
    k_small_gemm<<<(NGRAPH * HDIM + 255) / 256, 256, 0, stream>>>(z1, pw2, pb2, z2, NGRAPH, HDIM, HDIM, 1);
    k_small_gemm<<<(NGRAPH + 255) / 256, 256, 0, stream>>>(z2, pw3, pb3, out + nhd, NGRAPH, 1, HDIM, 0);
}
